// GINModel_39848706573591
// MI455X (gfx1250) — matrix operand order repaired and checked by reading, not compiled, not run on hardware
//
#include <hip/hip_runtime.h>
#include <hip/hip_bf16.h>

// ---------------------------------------------------------------------------
// GIN (2 layers) on MI455X / gfx1250, wave32 + V_WMMA_F32_16X16X32_BF16.
// Pipeline per call:
//   1) convert 4 weight matrices f32 -> bf16, transposed to [Nout][K]
//   2) agg = 0 ; scatter_add(agg[dst] += x[src])            (fp32 atomics, L2)
//   3) A_bf16 = bf16((1+eps0)*x + agg)                      (fused)
//   4) mid_bf16 = relu(A @ W1_0 + b1_0)                     (WMMA, bf16 out)
//   5) h0_f32   = relu(mid @ W2_0 + b2_0)                   (WMMA, f32 out)
//   6) agg = 0 ; scatter_add(agg[dst] += h0[src])
//   7) A_bf16 = bf16((1+eps1)*h0 + agg)
//   8) mid_bf16 = relu(A @ W1_1 + b1_1)                     (WMMA)
//   9) d_out    = mid @ W2_1 + b2_1                         (WMMA, logits f32)
//  10) d_out    = log_softmax(d_out) row-wise (wave32 shfl reductions)
// ---------------------------------------------------------------------------

typedef __bf16 bf16x16 __attribute__((ext_vector_type(16)));
typedef float  f32x8   __attribute__((ext_vector_type(8)));

__device__ __forceinline__ unsigned short f2bf(float f) {
    unsigned u = __float_as_uint(f);
    // round-to-nearest-even bf16 truncation
    unsigned r = (u + 0x7FFFu + ((u >> 16) & 1u)) >> 16;
    return (unsigned short)r;
}

// ---------------------------------------------------------------------------
// Weight convert + transpose: W[K][Nd] f32  ->  WT[Nd][K] bf16   (K == 256)
// ---------------------------------------------------------------------------
__global__ __launch_bounds__(256) void wtrans_kernel(
    const float* __restrict__ W, unsigned short* __restrict__ WT, int Nd) {
    int idx = blockIdx.x * 256 + threadIdx.x;       // idx = n*256 + k
    if (idx >= Nd * 256) return;
    int n = idx >> 8;
    int k = idx & 255;
    WT[idx] = f2bf(W[(size_t)k * Nd + n]);
}

// ---------------------------------------------------------------------------
// Edge scatter-add: agg[dst][c] += x[src][c].  4 edges per 256-thread block,
// 64 lanes * float4 = 256 channels per edge.  fp32 atomics hit L2.
// ---------------------------------------------------------------------------
__global__ __launch_bounds__(256) void scatter_add_kernel(
    const float* __restrict__ x, const long long* __restrict__ ei,
    float* __restrict__ agg, int E_) {
    int t  = blockIdx.x * 256 + threadIdx.x;
    int e  = t >> 6;
    int ch = (t & 63) * 4;
    if (e >= E_) return;
    long long s = ei[e];                 // edge_index[0][e]
    long long d = ei[(size_t)E_ + e];    // edge_index[1][e]
    float4 v = *(const float4*)(x + s * 256 + ch);
    float* p = agg + d * 256 + ch;
    __hip_atomic_fetch_add(p + 0, v.x, __ATOMIC_RELAXED, __HIP_MEMORY_SCOPE_AGENT);
    __hip_atomic_fetch_add(p + 1, v.y, __ATOMIC_RELAXED, __HIP_MEMORY_SCOPE_AGENT);
    __hip_atomic_fetch_add(p + 2, v.z, __ATOMIC_RELAXED, __HIP_MEMORY_SCOPE_AGENT);
    __hip_atomic_fetch_add(p + 3, v.w, __ATOMIC_RELAXED, __HIP_MEMORY_SCOPE_AGENT);
}

// ---------------------------------------------------------------------------
// A_bf16[i] = bf16((1+eps)*x[i] + agg[i]),  vectorized x4
// ---------------------------------------------------------------------------
__global__ __launch_bounds__(256) void fuse_kernel(
    const float* __restrict__ x, const float* __restrict__ agg,
    const float* __restrict__ epsP, unsigned short* __restrict__ outb,
    long total) {
    long i = ((long)blockIdx.x * 256 + threadIdx.x) * 4;
    if (i >= total) return;
    float c = 1.0f + *epsP;
    float4 xv = *(const float4*)(x + i);
    float4 av = *(const float4*)(agg + i);
    ushort4 o;
    o.x = f2bf(fmaf(c, xv.x, av.x));
    o.y = f2bf(fmaf(c, xv.y, av.y));
    o.z = f2bf(fmaf(c, xv.z, av.z));
    o.w = f2bf(fmaf(c, xv.w, av.w));
    *(ushort4*)(outb + i) = o;
}

// ---------------------------------------------------------------------------
// WMMA GEMM: C[M][Nd] = act(A[M][256] @ WT[Nd][256]^T + bias)
// Block = 256 threads = 8 waves (2 col-waves x 4 row-waves), 128x128 tile.
// Each wave: 32x64 = 2x4 tiles of 16x16, K-loop of 8 steps of 32.
// ---------------------------------------------------------------------------
template <int RELU, int OUTBF>
__global__ __launch_bounds__(256) void gemm_wmma_kernel(
    const unsigned short* __restrict__ A,    // bf16 bits [M][256]
    const unsigned short* __restrict__ WT,   // bf16 bits [Nd][256]
    const float* __restrict__ bias,
    void* __restrict__ Cout, int M, int Nd) {
    const int lane  = threadIdx.x & 31;
    const int wid   = threadIdx.x >> 5;
    const int waveM = wid & 3;            // 0..3  (rows)
    const int waveN = wid >> 2;           // 0..1  (cols)
    const int half  = lane >> 4;          // 0/1
    const int l16   = lane & 15;

    const int rowBase = blockIdx.x * 128 + waveM * 32;
    const int colBase = blockIdx.y * 128 + waveN * 64;

    f32x8 acc[2][4];
#pragma unroll
    for (int tm = 0; tm < 2; ++tm)
#pragma unroll
        for (int tn = 0; tn < 4; ++tn) acc[tm][tn] = f32x8{};

    // A rows this lane supplies (clamped so EXEC stays all-ones in K loop)
    int rowA[2];
#pragma unroll
    for (int tm = 0; tm < 2; ++tm) {
        int r = rowBase + tm * 16 + l16;
        rowA[tm] = r < M ? r : (M - 1);
    }

    union Frag { uint4 u[2]; bf16x16 v; };

#pragma unroll 2
    for (int k0 = 0; k0 < 256; k0 += 32) {
        Frag fa[2], fb[4];
        // A 16x32 bf16 fragment: regs 0-3 <- K[half*8 .. +7], regs 4-7 <- +16
#pragma unroll
        for (int tm = 0; tm < 2; ++tm) {
            const uint4* p =
                (const uint4*)(A + (size_t)rowA[tm] * 256 + k0 + half * 8);
            fa[tm].u[0] = p[0];   // 8 bf16
            fa[tm].u[1] = p[2];   // +16 elements = +2 uint4
        }
        // B 32x16 bf16 fragment from transposed weights: regs 0-3 <- K[half*8 .. +7], regs 4-7 <- +16
#pragma unroll
        for (int tn = 0; tn < 4; ++tn) {
            int n = colBase + tn * 16 + l16;
            const uint4* q =
                (const uint4*)(WT + (size_t)n * 256 + k0 + half * 8);
            fb[tn].u[0] = q[0];
            fb[tn].u[1] = q[2];
        }
#pragma unroll
        for (int tm = 0; tm < 2; ++tm)
#pragma unroll
            for (int tn = 0; tn < 4; ++tn)
                acc[tm][tn] = __builtin_amdgcn_wmma_f32_16x16x32_bf16(
                    false, fa[tm].v, false, fb[tn].v,
                    (short)0, acc[tm][tn], false, false);
    }

    // Epilogue: bias + optional ReLU, f32 or bf16 store.
    // C layout: reg r, lane -> (m = r + half*8, n = l16)
#pragma unroll
    for (int tm = 0; tm < 2; ++tm) {
        int rBaseT = rowBase + tm * 16 + half * 8;
#pragma unroll
        for (int tn = 0; tn < 4; ++tn) {
            int col  = colBase + tn * 16 + l16;
            float bv = bias[col];
#pragma unroll
            for (int r = 0; r < 8; ++r) {
                int ro = rBaseT + r;
                if (ro < M) {
                    float v = acc[tm][tn][r] + bv;
                    if (RELU) v = fmaxf(v, 0.0f);
                    if (OUTBF)
                        ((unsigned short*)Cout)[(size_t)ro * Nd + col] = f2bf(v);
                    else
                        ((float*)Cout)[(size_t)ro * Nd + col] = v;
                }
            }
        }
    }
}

// ---------------------------------------------------------------------------
// Row-wise log_softmax over 128 cols, in place. One wave32 per row.
// ---------------------------------------------------------------------------
__global__ __launch_bounds__(256) void logsoftmax_kernel(float* __restrict__ out,
                                                         int M) {
    int wid  = threadIdx.x >> 5;
    int lane = threadIdx.x & 31;
    int row  = blockIdx.x * 8 + wid;
    if (row >= M) return;   // uniform per wave
    float4 v = ((const float4*)(out + (size_t)row * 128))[lane];
    float mx = fmaxf(fmaxf(v.x, v.y), fmaxf(v.z, v.w));
#pragma unroll
    for (int o = 16; o; o >>= 1) mx = fmaxf(mx, __shfl_xor(mx, o, 32));
    float s = expf(v.x - mx) + expf(v.y - mx) + expf(v.z - mx) + expf(v.w - mx);
#pragma unroll
    for (int o = 16; o; o >>= 1) s += __shfl_xor(s, o, 32);
    float l = mx + logf(s);
    float4 r = make_float4(v.x - l, v.y - l, v.z - l, v.w - l);
    ((float4*)(out + (size_t)row * 128))[lane] = r;
}

// ---------------------------------------------------------------------------
extern "C" void kernel_launch(void* const* d_in, const int* in_sizes, int n_in,
                              void* d_out, int out_size, void* d_ws,
                              size_t ws_size, hipStream_t stream) {
    (void)in_sizes; (void)n_in; (void)out_size; (void)ws_size;
    const int NN = 50000, EE = 800000, C = 256, OUTC = 128;

    const float*     x    = (const float*)d_in[0];
    const long long* ei   = (const long long*)d_in[1];
    const float*     eps0 = (const float*)d_in[2];
    const float*     W1_0 = (const float*)d_in[3];
    const float*     b1_0 = (const float*)d_in[4];
    const float*     W2_0 = (const float*)d_in[5];
    const float*     b2_0 = (const float*)d_in[6];
    const float*     eps1 = (const float*)d_in[7];
    const float*     W1_1 = (const float*)d_in[8];
    const float*     b1_1 = (const float*)d_in[9];
    const float*     W2_1 = (const float*)d_in[10];
    const float*     b2_1 = (const float*)d_in[11];

    char*  ws  = (char*)d_ws;
    size_t off = 0;
    auto alloc = [&](size_t bytes) -> void* {
        void* p = ws + off;
        off += (bytes + 255) & ~(size_t)255;
        return p;
    };
    float*          agg  = (float*)alloc((size_t)NN * C * 4);
    unsigned short* Abf  = (unsigned short*)alloc((size_t)NN * C * 2);
    unsigned short* Mbf  = (unsigned short*)alloc((size_t)NN * C * 2);
    float*          h0   = (float*)alloc((size_t)NN * C * 4);
    unsigned short* WT10 = (unsigned short*)alloc((size_t)C * C * 2);
    unsigned short* WT20 = (unsigned short*)alloc((size_t)C * C * 2);
    unsigned short* WT11 = (unsigned short*)alloc((size_t)C * C * 2);
    unsigned short* WT21 = (unsigned short*)alloc((size_t)C * OUTC * 2);

    const int gridScatter = (EE * 64 + 255) / 256;          // 200000
    const int gridFuse    = ((long)NN * C / 4 + 255) / 256; // 12500
    const int gridMB      = (NN + 127) / 128;               // 391
    const int gridLSM     = (NN + 7) / 8;                   // 6250

    // weights -> bf16, transposed
    wtrans_kernel<<<(C * C + 255) / 256, 256, 0, stream>>>(W1_0, WT10, C);
    wtrans_kernel<<<(C * C + 255) / 256, 256, 0, stream>>>(W2_0, WT20, C);
    wtrans_kernel<<<(C * C + 255) / 256, 256, 0, stream>>>(W1_1, WT11, C);
    wtrans_kernel<<<(C * OUTC + 255) / 256, 256, 0, stream>>>(W2_1, WT21, OUTC);

    // ---- layer 0 ----
    hipMemsetAsync(agg, 0, (size_t)NN * C * 4, stream);
    scatter_add_kernel<<<gridScatter, 256, 0, stream>>>(x, ei, agg, EE);
    fuse_kernel<<<gridFuse, 256, 0, stream>>>(x, agg, eps0, Abf, (long)NN * C);
    gemm_wmma_kernel<1, 1><<<dim3(gridMB, 2), 256, 0, stream>>>(
        Abf, WT10, b1_0, Mbf, NN, C);
    gemm_wmma_kernel<1, 0><<<dim3(gridMB, 2), 256, 0, stream>>>(
        Mbf, WT20, b2_0, h0, NN, C);

    // ---- layer 1 ----
    hipMemsetAsync(agg, 0, (size_t)NN * C * 4, stream);
    scatter_add_kernel<<<gridScatter, 256, 0, stream>>>(h0, ei, agg, EE);
    fuse_kernel<<<gridFuse, 256, 0, stream>>>(h0, agg, eps1, Abf, (long)NN * C);
    gemm_wmma_kernel<1, 1><<<dim3(gridMB, 2), 256, 0, stream>>>(
        Abf, WT11, b1_1, Mbf, NN, C);
    gemm_wmma_kernel<0, 0><<<dim3(gridMB, 1), 256, 0, stream>>>(
        Mbf, WT21, b2_1, d_out, NN, OUTC);

    // ---- log_softmax ----
    logsoftmax_kernel<<<gridLSM, 256, 0, stream>>>((float*)d_out, NN);
}
